// GATLayer_36816459661559
// MI455X (gfx1250) — matrix operand order repaired and checked by reading, not compiled, not run on hardware
//
#include <hip/hip_runtime.h>

#define N_NODES 2048
#define N_EDGES 65536
#define D_IN    64
#define D_OUT   64
#define D_H     128   // 2*D_IN

typedef __attribute__((ext_vector_type(16))) _Float16 v16h;
typedef __attribute__((ext_vector_type(8)))  _Float16 v8h;
typedef __attribute__((ext_vector_type(8)))  float    v8f;

// ---------------------------------------------------------------- zero init
__global__ void zero_kernel(float* __restrict__ out, float* __restrict__ a_sum,
                            float* __restrict__ a_total) {
    int i = blockIdx.x * blockDim.x + threadIdx.x;
    if (i < N_NODES * D_OUT) out[i] = 0.0f;
    if (i < N_NODES)         a_sum[i] = 0.0f;
    if (i == 0)              *a_total = 0.0f;
}

// ------------------------------------------------- one-hot -> index extract
// src/tgt are [N, E] row-major; column e has exactly one 1.0 at row idx[e].
__global__ void extract_idx_kernel(const float* __restrict__ src,
                                   const float* __restrict__ tgt,
                                   int* __restrict__ sidx, int* __restrict__ tidx) {
    long i = (long)blockIdx.x * blockDim.x + threadIdx.x;   // group-of-4 id
    long l = i * 4;
    int n = (int)(l >> 16);          // E = 2^16
    int e = (int)(l & (N_EDGES - 1));
    float4 vs = ((const float4*)src)[i];
    float4 vt = ((const float4*)tgt)[i];
    if (vs.x > 0.5f) sidx[e + 0] = n;
    if (vs.y > 0.5f) sidx[e + 1] = n;
    if (vs.z > 0.5f) sidx[e + 2] = n;
    if (vs.w > 0.5f) sidx[e + 3] = n;
    if (vt.x > 0.5f) tidx[e + 0] = n;
    if (vt.y > 0.5f) tidx[e + 1] = n;
    if (vt.z > 0.5f) tidx[e + 2] = n;
    if (vt.w > 0.5f) tidx[e + 3] = n;
}

// --------------------------------------------------- f32 -> f16 conversion
__global__ void convert_kernel(const float* __restrict__ x, const float* __restrict__ W_f,
                               _Float16* __restrict__ x_h, _Float16* __restrict__ wf_h) {
    int i = blockIdx.x * blockDim.x + threadIdx.x;
    if (i < N_NODES * D_IN) x_h[i] = (_Float16)x[i];
    if (i < D_H * D_OUT)    wf_h[i] = (_Float16)W_f[i];
}

// ------------------------------------ attention logit a[e] + global sum
__global__ void edge_a_kernel(const float* __restrict__ x,
                              const int* __restrict__ sidx, const int* __restrict__ tidx,
                              const float* __restrict__ W_w, const float* __restrict__ b_w,
                              float* __restrict__ a_buf, float* __restrict__ a_total) {
    int e = blockIdx.x * blockDim.x + threadIdx.x;
    int s = sidx[e], t = tidx[e];
    float acc = b_w[0];
    #pragma unroll 8
    for (int k = 0; k < D_IN; ++k) acc += x[s * D_IN + k] * W_w[k];
    #pragma unroll 8
    for (int k = 0; k < D_IN; ++k) acc += x[t * D_IN + k] * W_w[D_IN + k];
    a_buf[e] = acc;
    __shared__ float s_acc;
    if (threadIdx.x == 0) s_acc = 0.0f;
    __syncthreads();
    atomicAdd(&s_acc, acc);
    __syncthreads();
    if (threadIdx.x == 0) atomicAdd(a_total, s_acc);
}

// -------------------------- a_exp = exp(a - mean); segment-sum per target
__global__ void edge_exp_kernel(const float* __restrict__ a_buf,
                                const int* __restrict__ tidx,
                                const float* __restrict__ a_total,
                                float* __restrict__ a_exp, float* __restrict__ a_sum) {
    int e = blockIdx.x * blockDim.x + threadIdx.x;
    float mean = (*a_total) * (1.0f / (float)N_EDGES);
    float ex = expf(a_buf[e] - mean);
    a_exp[e] = ex;
    atomicAdd(&a_sum[tidx[e]], ex);
}

// ----------------------------------------------------------- main WMMA GEMM
// Per wave: 16 edges; D = relu(h @ W_f + b) * att, scattered to out[tgt].
// h tile [16 x 128] staged in LDS (f16); W_f pre-swizzled in LDS so each
// lane's entire v16h B fragment is 32 contiguous bytes (2x ds_load_b128).
__global__ __launch_bounds__(256) void gat_wmma_kernel(
    const _Float16* __restrict__ x_h, const _Float16* __restrict__ wf_h,
    const int* __restrict__ sidx, const int* __restrict__ tidx,
    const float* __restrict__ a_exp, const float* __restrict__ a_sum,
    const float* __restrict__ b_f, float* __restrict__ out) {
    // lds_w[(((k*64+n)*2 + half)*16) + j] = W[32k + half*16 + j][n]
    __shared__ __align__(32) _Float16 lds_w[D_H * D_OUT];      // 16 KB, swizzled
    __shared__ __align__(32) _Float16 lds_h[8 * 16 * D_H];     // 32 KB
    __shared__ float lds_att[8 * 16];
    __shared__ int   lds_tgt[8 * 16];

    const int tid  = threadIdx.x;
    const int wave = tid >> 5;
    const int lane = tid & 31;
    const int edgeBase = (blockIdx.x * 8 + wave) * 16;

    // --- phase 1: cooperative LDS staging -------------------------------
    // W_f (f16) -> LDS, scattered into B-fragment order (once per block)
    for (int i = tid; i < D_H * D_OUT; i += 256) {
        int n    = i & 63;          // source column
        int r    = i >> 6;          // source row 0..127 (K index)
        int k    = r >> 5;          // 32-wide K block
        int half = (r >> 4) & 1;    // which 16-K half within the block
        int j    = r & 15;
        lds_w[((((k << 6) | n) << 1) | half) * 16 + j] = wf_h[i];
    }
    { // gather 16 edges x 128 feats (src||tgt) in 16-byte chunks
        uint4* hdst = (uint4*)&lds_h[wave * (16 * D_H)];
        for (int c = lane; c < 16 * 16; c += 32) {
            int e  = c >> 4;        // edge within tile
            int ch = c & 15;        // 16B chunk: 0..7 = src feats, 8..15 = tgt
            int node = (ch < 8) ? sidx[edgeBase + e] : tidx[edgeBase + e];
            int f = (ch & 7) * 8;   // feature offset (8 halves per chunk)
            hdst[e * 16 + ch] = *(const uint4*)&x_h[node * D_IN + f];
        }
    }
    if (lane < 16) { // per-edge softmax weight + target node
        int eg = edgeBase + lane;
        int tg = tidx[eg];
        lds_tgt[wave * 16 + lane] = tg;
        lds_att[wave * 16 + lane] = a_exp[eg] / (a_sum[tg] + 1e-6f);
    }
    __syncthreads();

    // --- phase 2: WMMA (EXEC uniform here) ------------------------------
    const int m    = lane & 15;
    const int half = lane >> 4;
    const _Float16* hb = &lds_h[wave * (16 * D_H)];

    // A fragments: 16x32 f16 layout; per lane: halves [32k+16h .. +7] and
    // [32k+16h+8 .. +7] -> two contiguous 16B chunks, concatenated.
    v16h afr[4];
    #pragma unroll
    for (int k = 0; k < 4; ++k) {
        const _Float16* ha = &hb[m * D_H + 32 * k + half * 16];
        v8h lo = *(const v8h*)(ha);
        v8h hi = *(const v8h*)(ha + 8);
        afr[k] = __builtin_shufflevector(lo, hi,
                                         0, 1, 2, 3, 4, 5, 6, 7,
                                         8, 9, 10, 11, 12, 13, 14, 15);
    }

    #pragma unroll
    for (int nb = 0; nb < 4; ++nb) {
        int n2 = nb * 16 + m;              // output column for this lane
        float bias = b_f[n2];
        v8f c = {0.f, 0.f, 0.f, 0.f, 0.f, 0.f, 0.f, 0.f};
        #pragma unroll
        for (int k = 0; k < 4; ++k) {
            // B fragment: 32 contiguous bytes in swizzled LDS
            v16h bfr = *(const v16h*)&lds_w[((((k << 6) | n2) << 1) | half) * 16];
            c = __builtin_amdgcn_wmma_f32_16x16x32_f16(
                    false, afr[k], false, bfr, (short)0, c, false, false);
        }
        // D layout: VGPR g -> M = g + 8*half, N = n2
        #pragma unroll
        for (int g = 0; g < 8; ++g) {
            int me = half * 8 + g;
            float v = c[g] + bias;
            v = v > 0.0f ? v : 0.0f;
            v *= lds_att[wave * 16 + me];
            atomicAdd(&out[lds_tgt[wave * 16 + me] * D_OUT + n2], v);
        }
    }
}

// ---------------------------------------------------------------- launcher
extern "C" void kernel_launch(void* const* d_in, const int* in_sizes, int n_in,
                              void* d_out, int out_size, void* d_ws, size_t ws_size,
                              hipStream_t stream) {
    const float* x   = (const float*)d_in[0];
    const float* src = (const float*)d_in[1];
    const float* tgt = (const float*)d_in[2];
    const float* W_f = (const float*)d_in[3];
    const float* b_f = (const float*)d_in[4];
    const float* W_w = (const float*)d_in[5];
    const float* b_w = (const float*)d_in[6];
    float* out = (float*)d_out;

    char* ws = (char*)d_ws;
    int*      sidx    = (int*)ws;      ws += (size_t)N_EDGES * 4;
    int*      tidx    = (int*)ws;      ws += (size_t)N_EDGES * 4;
    float*    a_buf   = (float*)ws;    ws += (size_t)N_EDGES * 4;
    float*    a_exp   = (float*)ws;    ws += (size_t)N_EDGES * 4;
    float*    a_sum   = (float*)ws;    ws += (size_t)N_NODES * 4;
    float*    a_total = (float*)ws;    ws += 32;
    _Float16* x_h     = (_Float16*)ws; ws += (size_t)N_NODES * D_IN * 2;
    _Float16* wf_h    = (_Float16*)ws; ws += (size_t)D_H * D_OUT * 2;

    zero_kernel<<<(N_NODES * D_OUT + 255) / 256, 256, 0, stream>>>(out, a_sum, a_total);

    long groups = (long)N_NODES * N_EDGES / 4;   // float4 per thread
    extract_idx_kernel<<<(unsigned int)(groups / 256), 256, 0, stream>>>(src, tgt, sidx, tidx);

    convert_kernel<<<(N_NODES * D_IN + 255) / 256, 256, 0, stream>>>(x, W_f, x_h, wf_h);

    edge_a_kernel<<<N_EDGES / 256, 256, 0, stream>>>(x, sidx, tidx, W_w, b_w, a_buf, a_total);

    edge_exp_kernel<<<N_EDGES / 256, 256, 0, stream>>>(a_buf, tidx, a_total, a_exp, a_sum);

    gat_wmma_kernel<<<N_EDGES / 128, 256, 0, stream>>>(x_h, wf_h, sidx, tidx,
                                                       a_exp, a_sum, b_f, out);
}
